// GIN_Classifier_266287972761
// MI455X (gfx1250) — matrix operand order repaired and checked by reading, not compiled, not run on hardware
//
#include <hip/hip_runtime.h>
#include <hip/hip_bf16.h>

typedef __attribute__((ext_vector_type(16))) _Float16 v16h;
typedef __attribute__((ext_vector_type(8)))  float    v8f;

#define DIMF 128
#define TS   132   // padded LDS row stride (floats): 132%64=4 -> conflict-free col reads, 132*4%16==0 -> b128 ok

__device__ __forceinline__ void atomAddG(float* p, float v) {
    (void)__hip_atomic_fetch_add(p, v, __ATOMIC_RELAXED, __HIP_MEMORY_SCOPE_AGENT);
}
__device__ __forceinline__ void atomAddL(float* p, float v) {
    (void)__hip_atomic_fetch_add(p, v, __ATOMIC_RELAXED, __HIP_MEMORY_SCOPE_WORKGROUP);
}

__device__ __forceinline__ v16h pack16(float4 a, float4 b, float4 c, float4 d) {
    v16h r;
    r[0]=(_Float16)a.x; r[1]=(_Float16)a.y; r[2]=(_Float16)a.z;  r[3]=(_Float16)a.w;
    r[4]=(_Float16)b.x; r[5]=(_Float16)b.y; r[6]=(_Float16)b.z;  r[7]=(_Float16)b.w;
    r[8]=(_Float16)c.x; r[9]=(_Float16)c.y; r[10]=(_Float16)c.z; r[11]=(_Float16)c.w;
    r[12]=(_Float16)d.x;r[13]=(_Float16)d.y;r[14]=(_Float16)d.z; r[15]=(_Float16)d.w;
    return r;
}

// ---------------- zero ----------------
__global__ void zero_f32(float* p, int n) {
    int i = blockIdx.x * 256 + threadIdx.x;
    if (i < n) p[i] = 0.0f;
}

// ------------- weight pack: w[128][128] f32 -> WMMA B-operand f16 tiles -------------
// layout: [(kt*8+nt)*32 + lane][16 halfs]; lane L: N = nt*16 + (L&15), K = kt*32 + ((L&16)?16:0) + j
__global__ __launch_bounds__(256) void pack_w(const float* __restrict__ w, _Float16* __restrict__ dst) {
    int t = blockIdx.x * 256 + threadIdx.x;      // 0..1023 = kt(4) x nt(8) x lane(32)
    if (t >= 1024) return;
    int lane = t & 31, nt = (t >> 5) & 7, kt = t >> 8;
    _Float16* o = dst + ((size_t)t << 4);
    int nc = nt * 16 + (lane & 15);
    int kb = kt * 32 + ((lane & 16) ? 16 : 0);
#pragma unroll
    for (int j = 0; j < 16; j++) o[j] = (_Float16)w[(size_t)(kb + j) * DIMF + nc];
}

// ------------- edge scatter-add: wave per edge, lane covers 4 floats -------------
__global__ __launch_bounds__(256) void gin_scatter(const float* __restrict__ xin,
                                                   const int* __restrict__ ei,
                                                   float* agg, int nE) {
    long t = (long)blockIdx.x * 256 + threadIdx.x;
    int e = (int)(t >> 5), lane = (int)(t & 31);
    if (e < nE) {
        int s = ei[e], d = ei[nE + e];
        float4 v = *(const float4*)(xin + (size_t)s * DIMF + lane * 4);
        float* o = agg + (size_t)d * DIMF + lane * 4;
        atomAddG(o + 0, v.x); atomAddG(o + 1, v.y);
        atomAddG(o + 2, v.z); atomAddG(o + 3, v.w);
    }
}

// ------------- fused GIN MLP: (x+agg) @ W1 +b1, relu, @ W2 +b2, relu, + BN stats -------------
// block = 128 threads = 4 waves; wave handles a 16-row tile; 64 WMMAs / wave
__global__ __launch_bounds__(128) void gin_mlp(const float* __restrict__ xin,
                                               const float* __restrict__ agg,
                                               const _Float16* __restrict__ pw1,
                                               const _Float16* __restrict__ pw2,
                                               const float* __restrict__ b1,
                                               const float* __restrict__ b2,
                                               float* hout,
                                               float* colsum, float* colsumsq,
                                               int nrows) {
    __shared__ float lt[4][16 * TS];
    __shared__ float lsum[DIMF];
    __shared__ float lsq[DIMF];
    int tid = threadIdx.x;
    if (tid < DIMF) { lsum[tid] = 0.0f; lsq[tid] = 0.0f; }
    __syncthreads();

    int wv = tid >> 5, lane = tid & 31;
    int m0 = (blockIdx.x * 4 + wv) * 16;
    if (m0 < nrows) {
        float* T = lt[wv];
        int lm = lane & 15;            // A: row M, C: col N
        int hi = (lane >> 4) & 1;      // lane half
        const float* xrow = xin + (size_t)(m0 + lm) * DIMF;
        const float* arow = agg + (size_t)(m0 + lm) * DIMF;
        const _Float16* pw1l = pw1 + ((size_t)lane << 4);
        const _Float16* pw2l = pw2 + ((size_t)lane << 4);

        // ---- GEMM1: acc[nt] = (x+agg) @ W1 ----
        v8f acc[8];
#pragma unroll
        for (int i = 0; i < 8; i++) acc[i] = (v8f){};
#pragma unroll
        for (int kt = 0; kt < 4; kt++) {
            int kb = kt * 32 + (hi ? 16 : 0);
            float4 x0 = *(const float4*)(xrow + kb);
            float4 x1 = *(const float4*)(xrow + kb + 4);
            float4 x2 = *(const float4*)(xrow + kb + 8);
            float4 x3 = *(const float4*)(xrow + kb + 12);
            float4 a0 = *(const float4*)(arow + kb);
            float4 a1 = *(const float4*)(arow + kb + 4);
            float4 a2 = *(const float4*)(arow + kb + 8);
            float4 a3 = *(const float4*)(arow + kb + 12);
            // preload all 8 B tiles of this k-slab (one clause, one wait)
            v16h Bt[8];
#pragma unroll
            for (int nt = 0; nt < 8; nt++)
                Bt[nt] = *(const v16h*)(pw1l + ((size_t)(kt * 8 + nt) << 9));
            float4 s0 = make_float4(x0.x+a0.x, x0.y+a0.y, x0.z+a0.z, x0.w+a0.w);
            float4 s1 = make_float4(x1.x+a1.x, x1.y+a1.y, x1.z+a1.z, x1.w+a1.w);
            float4 s2 = make_float4(x2.x+a2.x, x2.y+a2.y, x2.z+a2.z, x2.w+a2.w);
            float4 s3 = make_float4(x3.x+a3.x, x3.y+a3.y, x3.z+a3.z, x3.w+a3.w);
            v16h A = pack16(s0, s1, s2, s3);
#pragma unroll
            for (int nt = 0; nt < 8; nt++)
                acc[nt] = __builtin_amdgcn_wmma_f32_16x16x32_f16(
                    false, A, false, Bt[nt], (short)0, acc[nt], false, false);
        }

        // ---- bias + relu -> transpose through LDS ----
#pragma unroll
        for (int nt = 0; nt < 8; nt++) {
            float bias = b1[nt * 16 + lm];
#pragma unroll
            for (int v = 0; v < 8; v++) {
                float t = acc[nt][v] + bias;
                T[(v + 8 * hi) * TS + nt * 16 + lm] = t > 0.0f ? t : 0.0f;
            }
        }

        // ---- GEMM2: acc2[nt] = t @ W2 ----
        v8f acc2[8];
#pragma unroll
        for (int i = 0; i < 8; i++) acc2[i] = (v8f){};
        const float* R = T + lm * TS;
#pragma unroll
        for (int kt = 0; kt < 4; kt++) {
            int kb = kt * 32 + (hi ? 16 : 0);
            float4 t0 = *(const float4*)(R + kb);
            float4 t1 = *(const float4*)(R + kb + 4);
            float4 t2 = *(const float4*)(R + kb + 8);
            float4 t3 = *(const float4*)(R + kb + 12);
            v16h Bt[8];
#pragma unroll
            for (int nt = 0; nt < 8; nt++)
                Bt[nt] = *(const v16h*)(pw2l + ((size_t)(kt * 8 + nt) << 9));
            v16h A = pack16(t0, t1, t2, t3);
#pragma unroll
            for (int nt = 0; nt < 8; nt++)
                acc2[nt] = __builtin_amdgcn_wmma_f32_16x16x32_f16(
                    false, A, false, Bt[nt], (short)0, acc2[nt], false, false);
        }

        // ---- bias + relu, BN partial stats, stage into LDS (row-major) ----
#pragma unroll
        for (int nt = 0; nt < 8; nt++) {
            float bias = b2[nt * 16 + lm];
            float s = 0.0f, q = 0.0f;
#pragma unroll
            for (int v = 0; v < 8; v++) {
                float h = acc2[nt][v] + bias;
                h = h > 0.0f ? h : 0.0f;
                T[(v + 8 * hi) * TS + nt * 16 + lm] = h;
                s += h; q += h * h;
            }
            atomAddL(&lsum[nt * 16 + lm], s);
            atomAddL(&lsq[nt * 16 + lm], q);
        }
        // ---- contiguous b128 row stores: one full 512B row per instruction ----
        float* orow = hout + (size_t)m0 * DIMF + lane * 4;
#pragma unroll
        for (int r = 0; r < 16; r++) {
            float4 v = *(const float4*)(T + r * TS + lane * 4);
            *(float4*)(orow + (size_t)r * DIMF) = v;
        }
    }
    __syncthreads();
    if (tid < DIMF) {
        atomAddG(&colsum[tid], lsum[tid]);
        atomAddG(&colsumsq[tid], lsq[tid]);
    }
}

// ------------- BN finalize -------------
__global__ void gin_bnstats(const float* cs, const float* cq,
                            const float* gamma, const float* beta,
                            float* scale, float* shift, float invN) {
    int c = threadIdx.x;
    float mu  = cs[c] * invN;
    float var = cq[c] * invN - mu * mu;
    float inv = rsqrtf(var + 1e-5f);
    float sc  = gamma[c] * inv;
    scale[c] = sc;
    shift[c] = beta[c] - mu * sc;
}

// ------------- BN apply (in place) -------------
__global__ __launch_bounds__(256) void gin_bnorm(float* h, const float* __restrict__ scale,
                                                 const float* __restrict__ shift, int n32) {
    int t = blockIdx.x * 256 + threadIdx.x;
    if (t < n32) {
        int c = (t & 31) * 4;
        float4* p = (float4*)h + t;
        float4 v = *p;
        v.x = v.x * scale[c + 0] + shift[c + 0];
        v.y = v.y * scale[c + 1] + shift[c + 1];
        v.z = v.z * scale[c + 2] + shift[c + 2];
        v.w = v.w * scale[c + 3] + shift[c + 3];
        *p = v;
    }
}

// ------------- global_add_pool -------------
__global__ __launch_bounds__(256) void gin_pool(const float* __restrict__ h,
                                                const int* __restrict__ batch,
                                                float* g, int nN) {
    long t = (long)blockIdx.x * 256 + threadIdx.x;
    int node = (int)(t >> 5), lane = (int)(t & 31);
    if (node < nN) {
        int b = batch[node];
        float4 v = *(const float4*)(h + (size_t)node * DIMF + lane * 4);
        float* o = g + (size_t)b * DIMF + lane * 4;
        atomAddG(o + 0, v.x); atomAddG(o + 1, v.y);
        atomAddG(o + 2, v.z); atomAddG(o + 3, v.w);
    }
}

// ------------- fc1+relu, fc2, log_softmax (tiny: VALU) -------------
__global__ __launch_bounds__(128) void gin_head(const float* __restrict__ g,
                                                const float* __restrict__ w1,
                                                const float* __restrict__ b1,
                                                const float* __restrict__ w2,
                                                const float* __restrict__ b2,
                                                float* __restrict__ out) {
    __shared__ float gr[DIMF];
    __shared__ float y[DIMF];
    __shared__ float zl[16];
    int tid = threadIdx.x, gid = blockIdx.x;
    gr[tid] = g[(size_t)gid * DIMF + tid];
    __syncthreads();
    float acc = b1[tid];
#pragma unroll 8
    for (int k = 0; k < DIMF; k++) acc += gr[k] * w1[(size_t)k * DIMF + tid];
    y[tid] = acc > 0.0f ? acc : 0.0f;
    __syncthreads();
    if (tid < 10) {
        float z = b2[tid];
#pragma unroll 8
        for (int j = 0; j < DIMF; j++) z += y[j] * w2[(size_t)j * 10 + tid];
        zl[tid] = z;
    }
    __syncthreads();
    if (tid < 10) {
        float m = zl[0];
#pragma unroll
        for (int c = 1; c < 10; c++) m = fmaxf(m, zl[c]);
        float se = 0.0f;
#pragma unroll
        for (int c = 0; c < 10; c++) se += expf(zl[c] - m);
        out[(size_t)gid * 10 + tid] = zl[tid] - m - logf(se);
    }
}

extern "C" void kernel_launch(void* const* d_in, const int* in_sizes, int n_in,
                              void* d_out, int out_size, void* d_ws, size_t ws_size,
                              hipStream_t stream) {
    const float* x     = (const float*)d_in[0];
    const int*   ei    = (const int*)d_in[1];
    const int*   batch = (const int*)d_in[2];
    const float* fc1_b = (const float*)d_in[3];
    const float* fc1_w = (const float*)d_in[4];
    const float* fc2_b = (const float*)d_in[5];
    const float* fc2_w = (const float*)d_in[6];

    const int N = in_sizes[0] / DIMF;
    const int E = in_sizes[1] / 2;
    const int G = out_size / 10;

    // workspace layout
    char* w = (char*)d_ws;
    float*     hbuf   = (float*)w;                                        // N*128 f32
    float*     aggb   = hbuf + (size_t)N * DIMF;                          // N*128 f32
    _Float16*  packed = (_Float16*)(w + (size_t)N * DIMF * 8);            // 10*16384 f16
    float*     stats  = (float*)((char*)packed + (size_t)10 * 16384 * 2); // 512 f32
    float*     colsum = stats, *colsumsq = stats + 128;
    float*     scale  = stats + 256, *shift = stats + 384;
    float*     gpool  = stats + 512;                                      // G*128 f32

    // pack all 10 weight matrices into WMMA B-operand layout
    for (int l = 0; l < 5; l++) {
        const float* w1 = (const float*)d_in[7 + 6 * l + 4];
        const float* w2 = (const float*)d_in[7 + 6 * l + 5];
        pack_w<<<4, 256, 0, stream>>>(w1, packed + (size_t)(2 * l) * 16384);
        pack_w<<<4, 256, 0, stream>>>(w2, packed + (size_t)(2 * l + 1) * 16384);
    }

    const int tiles = (N + 15) / 16;
    const int mlpBlocks = (tiles + 3) / 4;
    const int nd = N * DIMF;

    for (int l = 0; l < 5; l++) {
        const float* b1    = (const float*)d_in[7 + 6 * l + 0];
        const float* b2    = (const float*)d_in[7 + 6 * l + 1];
        const float* beta  = (const float*)d_in[7 + 6 * l + 2];
        const float* gamma = (const float*)d_in[7 + 6 * l + 3];
        const float* xin   = (l == 0) ? x : hbuf;

        zero_f32<<<(nd + 255) / 256, 256, 0, stream>>>(aggb, nd);
        zero_f32<<<1, 256, 0, stream>>>(colsum, 256);
        gin_scatter<<<(E + 7) / 8, 256, 0, stream>>>(xin, ei, aggb, E);
        gin_mlp<<<mlpBlocks, 128, 0, stream>>>(xin, aggb,
                                               packed + (size_t)(2 * l) * 16384,
                                               packed + (size_t)(2 * l + 1) * 16384,
                                               b1, b2, hbuf, colsum, colsumsq, N);
        gin_bnstats<<<1, 128, 0, stream>>>(colsum, colsumsq, gamma, beta, scale, shift, 1.0f / (float)N);
        gin_bnorm<<<(N * 32 + 255) / 256, 256, 0, stream>>>(hbuf, scale, shift, N * 32);
    }

    zero_f32<<<(G * DIMF + 255) / 256, 256, 0, stream>>>(gpool, G * DIMF);
    gin_pool<<<(N * 32 + 255) / 256, 256, 0, stream>>>(hbuf, batch, gpool, N);
    gin_head<<<G, 128, 0, stream>>>(gpool, fc1_w, fc1_b, fc2_w, fc2_b, (float*)d_out);
}
